// Attention_53068615909948
// MI455X (gfx1250) — matrix operand order repaired and checked by reading, not compiled, not run on hardware
//
#include <hip/hip_runtime.h>
#include <hip/hip_bf16.h>

typedef __attribute__((ext_vector_type(16))) _Float16 v16h;
typedef __attribute__((ext_vector_type(8)))  _Float16 v8h;
typedef __attribute__((ext_vector_type(4)))  _Float16 v4h;
typedef __attribute__((ext_vector_type(4)))  float    v4f;
typedef __attribute__((ext_vector_type(8)))  float    v8f;

#define BATCH 4
#define CCH   512
#define NSEQ  2048
#define HEADS 8
#define DHEAD 64

static __device__ __forceinline__ int lane_id() { return threadIdx.x & 31; }

// ---- WMMA fragment loaders (CDNA5 16x16x32 f16 layouts, wave32) ----

// A fragment: 16(M) x 32(K) f16 from row-major (K contiguous) source, leading dim ld.
// Lane L: row = L&15, half = L>>4; K = half*16+[0..15] contiguous. Two b128 loads.
static __device__ __forceinline__ v16h load_a_h(const _Float16* __restrict__ src, int ld, int k0) {
  int lane = lane_id();
  int row = lane & 15, half = lane >> 4;
  const _Float16* p = src + (size_t)row * ld + k0 + half * 16;
  v8h lo = *(const v8h*)(p);
  v8h hi = *(const v8h*)(p + 8);
  v16h a;
#pragma unroll
  for (int i = 0; i < 8; ++i) { a[i] = lo[i]; a[8 + i] = hi[i]; }
  return a;
}

// B fragment: 32(K) x 16(N) f16 from COLUMN-major storage (K contiguous per column),
// leading dim ld. Lane L: col = L&15, K = k0+(L>>4)*16+[0..15] contiguous -> two b128 loads.
static __device__ __forceinline__ v16h load_b_c(const _Float16* __restrict__ src, int ld,
                                                int k0, int col0) {
  int lane = lane_id();
  int col = lane & 15;
  int kb = k0 + (lane >> 4) * 16;
  const _Float16* p = src + (size_t)(col0 + col) * ld + kb;
  v8h lo = *(const v8h*)(p);
  v8h hi = *(const v8h*)(p + 8);
  v16h b;
#pragma unroll
  for (int i = 0; i < 8; ++i) { b[i] = lo[i]; b[8 + i] = hi[i]; }
  return b;
}

static __device__ __forceinline__ v8f wmma_f16(v16h a, v16h b, v8f c) {
  return __builtin_amdgcn_wmma_f32_16x16x32_f16(false, a, false, b, (short)0, c, false, false);
}

static __device__ __forceinline__ v8h cvt_h8(v8f a, float s) {
  v8h r;
#pragma unroll
  for (int i = 0; i < 8; ++i) r[i] = (_Float16)(a[i] * s);
  return r;
}

// ---- Kernel 0: one-time f32 -> f16 weight conversion ----
__global__ __launch_bounds__(256) void cvtw_kernel(const float* __restrict__ w,
                                                   _Float16* __restrict__ wh, int n4) {
  int i = blockIdx.x * 256 + threadIdx.x;
  if (i < n4) {
    v4f v = *(const v4f*)(w + (size_t)i * 4);
    v4h r;
#pragma unroll
    for (int k = 0; k < 4; ++k) r[k] = (_Float16)v[k];
    *(v4h*)(wh + (size_t)i * 4) = r;
  }
}

// ---- Kernel 1: channel LayerNorm; f32 [b][c][n] -> f16 [b][n][c] via LDS transpose ----
__global__ __launch_bounds__(256) void ln_kernel(const float* __restrict__ x,
                                                 const float* __restrict__ g,
                                                 _Float16* __restrict__ xn) {
  int tx = threadIdx.x, ty = threadIdx.y;          // block (32, 8); 32 positions per block
  int bb = blockIdx.x >> 6;
  int n0 = (blockIdx.x & 63) * 32;
  const float* xb = x + (size_t)bb * CCH * NSEQ + n0 + tx;
  float s = 0.f, ss = 0.f;
#pragma unroll 4
  for (int c = ty; c < CCH; c += 8) {
    float v = xb[(size_t)c * NSEQ];
    s += v; ss += v * v;
  }
  __shared__ float rs[8][32], rq[8][32], mrow[32], rrow[32];
  rs[ty][tx] = s; rq[ty][tx] = ss;
  __syncthreads();
  if (ty == 0) {
    float a = 0.f, b2 = 0.f;
#pragma unroll
    for (int k = 0; k < 8; ++k) { a += rs[k][tx]; b2 += rq[k][tx]; }
    float mean = a * (1.0f / CCH);
    float var = b2 * (1.0f / CCH) - mean * mean;
    mrow[tx] = mean;
    rrow[tx] = rsqrtf(var + 1e-5f);
  }
  __syncthreads();
  float mean = mrow[tx], rstd = rrow[tx];
  __shared__ _Float16 tile[32][CCH + 8];           // +8 f16 pad: rows stay 16B-aligned
#pragma unroll 4
  for (int c = ty; c < CCH; c += 8) {
    float v = xb[(size_t)c * NSEQ];
    tile[tx][c] = (_Float16)((v - mean) * rstd * g[c]);
  }
  __syncthreads();
  int tid = ty * 32 + tx;
  _Float16* xo = xn + ((size_t)bb * NSEQ + n0) * CCH;
#pragma unroll
  for (int sgrp = 0; sgrp < 8; ++sgrp) {
    int chunk = tid + 256 * sgrp;                  // 2048 chunks of 8 f16
    int p = chunk >> 6, cc = (chunk & 63) * 8;
    v8h val = *(const v8h*)&tile[p][cc];
    *(v8h*)(xo + (size_t)p * CCH + cc) = val;
  }
}

// ---- Kernel 2: QKV GEMM; 32(M) x 64(N) register tile per wave, f16 weights ----
__global__ __launch_bounds__(256) void qkv_kernel(const _Float16* __restrict__ wq,  // f16 [3c][c]
                                                  const _Float16* __restrict__ xn,  // [b][n][c]
                                                  _Float16* __restrict__ qs,   // [bh][n][d] scaled
                                                  _Float16* __restrict__ kT,   // [bh][n][d]
                                                  _Float16* __restrict__ vv) { // [bh][d][n]
  int wid = blockIdx.x * 8 + (threadIdx.x >> 5);   // 4*48*32 waves
  int bb = wid / (48 * 32);
  int r  = wid % (48 * 32);
  int mg = r / 32, ng = r % 32;                    // 32-row group, 64-col group
  const _Float16* wa = wq + (size_t)(mg * 32) * CCH;
  const _Float16* xb = xn + (size_t)bb * NSEQ * CCH;
  v8f acc0[4] = {}, acc1[4] = {};
#pragma unroll 4
  for (int k0 = 0; k0 < CCH; k0 += 32) {
    v16h a0 = load_a_h(wa, CCH, k0);
    v16h a1 = load_a_h(wa + (size_t)16 * CCH, CCH, k0);
#pragma unroll
    for (int t = 0; t < 4; ++t) {
      v16h b = load_b_c(xb, CCH, k0, ng * 64 + t * 16);
      acc0[t] = wmma_f16(a0, b, acc0[t]);
      acc1[t] = wmma_f16(a1, b, acc1[t]);
    }
  }
  int lane = lane_id();
  int col = lane & 15, rowoff = (lane >> 4) * 8;
#pragma unroll
  for (int u = 0; u < 2; ++u) {
    v8f* acc = u ? acc1 : acc0;
    int ob = mg * 32 + u * 16;                     // tile base output-channel
    int sect = ob >> 9;                            // 0=q,1=k,2=v (uniform per tile)
    int hh = (ob & 511) >> 6;
    int dd0 = (ob & 63) + rowoff;
    size_t bh = (size_t)(bb * HEADS + hh);
#pragma unroll
    for (int t = 0; t < 4; ++t) {
      int j = ng * 64 + t * 16 + col;
      if (sect == 0) {
        *(v8h*)(qs + (bh * NSEQ + j) * DHEAD + dd0) = cvt_h8(acc[t], 0.125f);
      } else if (sect == 1) {
        *(v8h*)(kT + (bh * NSEQ + j) * DHEAD + dd0) = cvt_h8(acc[t], 1.0f);
      } else {
#pragma unroll
        for (int v = 0; v < 8; ++v)
          vv[(bh * DHEAD + dd0 + v) * NSEQ + j] = (_Float16)acc[t][v];
      }
    }
  }
}

// ---- Kernel 3: flash attention; one wave per (b,h, 32 query columns) ----
// S^T = K^T @ Q_scaled (rows=j, cols=i): softmax stats are per-lane scalars.
// Two i-tiles per wave share the K^T and V A-fragments -> 1 vmem per wmma.
__global__ __launch_bounds__(256) void attn_kernel(const _Float16* __restrict__ qs,
                                                   const _Float16* __restrict__ kT,
                                                   const _Float16* __restrict__ vv,
                                                   _Float16* __restrict__ ao) { // [b][n][c]
  int wid = blockIdx.x * 8 + (threadIdx.x >> 5);   // 4*8*64 waves
  int it = wid & 63;
  int bh = wid >> 6;
  int i0 = it * 32;
  const _Float16* qb = qs + (size_t)bh * NSEQ * DHEAD;
  const _Float16* kb = kT + (size_t)bh * NSEQ * DHEAD;
  const _Float16* vb = vv + (size_t)bh * DHEAD * NSEQ;

  v16h bqA0 = load_b_c(qb, DHEAD, 0, i0);
  v16h bqA1 = load_b_c(qb, DHEAD, 32, i0);
  v16h bqB0 = load_b_c(qb, DHEAD, 0, i0 + 16);
  v16h bqB1 = load_b_c(qb, DHEAD, 32, i0 + 16);

  v8f accA[4] = {}, accB[4] = {};                  // out^T tiles: rows d, cols i
  float mA = -3.0e38f, lA = 0.f, mB = -3.0e38f, lB = 0.f;
  int lane = lane_id();
  bool lo = (lane < 16);

  for (int j0 = 0; j0 < NSEQ; j0 += 32) {
    v16h ak00 = load_a_h(kb + (size_t)j0 * DHEAD, DHEAD, 0);
    v16h ak01 = load_a_h(kb + (size_t)j0 * DHEAD, DHEAD, 32);
    v16h ak10 = load_a_h(kb + (size_t)(j0 + 16) * DHEAD, DHEAD, 0);
    v16h ak11 = load_a_h(kb + (size_t)(j0 + 16) * DHEAD, DHEAD, 32);
    v8f sA0 = {}, sA1 = {}, sB0 = {}, sB1 = {};
    sA0 = wmma_f16(ak00, bqA0, sA0); sA0 = wmma_f16(ak01, bqA1, sA0);
    sA1 = wmma_f16(ak10, bqA0, sA1); sA1 = wmma_f16(ak11, bqA1, sA1);
    sB0 = wmma_f16(ak00, bqB0, sB0); sB0 = wmma_f16(ak01, bqB1, sB0);
    sB1 = wmma_f16(ak10, bqB0, sB1); sB1 = wmma_f16(ak11, bqB1, sB1);

    float mtA = -3.0e38f, mtB = -3.0e38f;
#pragma unroll
    for (int v = 0; v < 8; ++v) {
      mtA = fmaxf(mtA, fmaxf(sA0[v], sA1[v]));
      mtB = fmaxf(mtB, fmaxf(sB0[v], sB1[v]));
    }
    mtA = fmaxf(mtA, __shfl_xor(mtA, 16));
    mtB = fmaxf(mtB, __shfl_xor(mtB, 16));
    float mnA = fmaxf(mA, mtA), mnB = fmaxf(mB, mtB);
    float alA = __expf(mA - mnA), alB = __expf(mB - mnB);
    float pA0[8], pA1[8], pB0[8], pB1[8], rsA = 0.f, rsB = 0.f;
#pragma unroll
    for (int v = 0; v < 8; ++v) {
      pA0[v] = __expf(sA0[v] - mnA); pA1[v] = __expf(sA1[v] - mnA);
      pB0[v] = __expf(sB0[v] - mnB); pB1[v] = __expf(sB1[v] - mnB);
      rsA += pA0[v] + pA1[v];
      rsB += pB0[v] + pB1[v];
    }
    rsA += __shfl_xor(rsA, 16);
    rsB += __shfl_xor(rsB, 16);
    lA = lA * alA + rsA; mA = mnA;
    lB = lB * alB + rsB; mB = mnB;
#pragma unroll
    for (int v = 0; v < 8; ++v) {
#pragma unroll
      for (int t = 0; t < 4; ++t) { accA[t][v] *= alA; accB[t][v] *= alB; }
    }

    v16h pbA, pbB;
#pragma unroll
    for (int v = 0; v < 8; ++v) {
      float xA0 = __shfl_xor(pA0[v], 16), xA1 = __shfl_xor(pA1[v], 16);
      float xB0 = __shfl_xor(pB0[v], 16), xB1 = __shfl_xor(pB1[v], 16);
      pbA[v]     = (_Float16)(lo ? pA0[v] : xA1);
      pbA[8 + v] = (_Float16)(lo ? xA0 : pA1[v]);
      pbB[v]     = (_Float16)(lo ? pB0[v] : xB1);
      pbB[8 + v] = (_Float16)(lo ? xB0 : pB1[v]);
    }

#pragma unroll
    for (int t = 0; t < 4; ++t) {
      v16h av = load_a_h(vb + (size_t)(t * 16) * NSEQ, NSEQ, j0);
      accA[t] = wmma_f16(av, pbA, accA[t]);
      accB[t] = wmma_f16(av, pbB, accB[t]);
    }
  }

  float liA = 1.0f / lA, liB = 1.0f / lB;
  int col = lane & 15, rowoff = (lane >> 4) * 8;
  int bb = bh >> 3, hh = bh & 7;
  _Float16* obA = ao + ((size_t)bb * NSEQ + i0 + col) * CCH + hh * DHEAD + rowoff;
  _Float16* obB = obA + (size_t)16 * CCH;
#pragma unroll
  for (int t = 0; t < 4; ++t) {
    *(v8h*)(obA + t * 16) = cvt_h8(accA[t], liA);
    *(v8h*)(obB + t * 16) = cvt_h8(accB[t], liB);
  }
}

// ---- Kernel 4: output projection + bias; 32(M) x 64(N) register tile, f16 weights ----
__global__ __launch_bounds__(256) void proj_kernel(const _Float16* __restrict__ wo, // f16 [c][c]
                                                   const float* __restrict__ b_out,
                                                   const _Float16* __restrict__ ao, // [b][n][c]
                                                   float* __restrict__ out) {
  int wid = blockIdx.x * 8 + (threadIdx.x >> 5);   // 4*16*32 waves
  int bb = wid / (16 * 32);
  int r  = wid % (16 * 32);
  int mg = r / 32, ng = r % 32;
  const _Float16* wa = wo + (size_t)(mg * 32) * CCH;
  const _Float16* xb = ao + (size_t)bb * NSEQ * CCH;
  v8f acc0[4] = {}, acc1[4] = {};
#pragma unroll 4
  for (int k0 = 0; k0 < CCH; k0 += 32) {
    v16h a0 = load_a_h(wa, CCH, k0);
    v16h a1 = load_a_h(wa + (size_t)16 * CCH, CCH, k0);
#pragma unroll
    for (int t = 0; t < 4; ++t) {
      v16h b = load_b_c(xb, CCH, k0, ng * 64 + t * 16);
      acc0[t] = wmma_f16(a0, b, acc0[t]);
      acc1[t] = wmma_f16(a1, b, acc1[t]);
    }
  }
  int lane = lane_id();
  int col = lane & 15, rowoff = (lane >> 4) * 8;
#pragma unroll
  for (int u = 0; u < 2; ++u) {
    v8f* acc = u ? acc1 : acc0;
#pragma unroll
    for (int t = 0; t < 4; ++t) {
      int j = ng * 64 + t * 16 + col;
#pragma unroll
      for (int v = 0; v < 8; ++v) {
        int o = mg * 32 + u * 16 + rowoff + v;
        out[((size_t)bb * CCH + o) * NSEQ + j] = acc[t][v] + b_out[o];
      }
    }
  }
}

extern "C" void kernel_launch(void* const* d_in, const int* in_sizes, int n_in,
                              void* d_out, int out_size, void* d_ws, size_t ws_size,
                              hipStream_t stream) {
  const float* x     = (const float*)d_in[0];
  const float* g     = (const float*)d_in[1];
  const float* w_qkv = (const float*)d_in[2];
  const float* w_out = (const float*)d_in[3];
  const float* b_out = (const float*)d_in[4];
  float* out = (float*)d_out;

  size_t S = (size_t)BATCH * CCH * NSEQ;           // 4 Mi elements per activation buffer
  _Float16* xn = (_Float16*)d_ws;
  _Float16* qs = xn + S;
  _Float16* kT = qs + S;
  _Float16* vv = kT + S;
  _Float16* ao = vv + S;
  _Float16* wqh = ao + S;                          // 3c*c f16
  _Float16* woh = wqh + (size_t)3 * CCH * CCH;     // c*c f16

  cvtw_kernel<<<(3 * CCH * CCH / 4 + 255) / 256, 256, 0, stream>>>(w_qkv, wqh, 3 * CCH * CCH / 4);
  cvtw_kernel<<<(CCH * CCH / 4 + 255) / 256, 256, 0, stream>>>(w_out, woh, CCH * CCH / 4);
  dim3 lnb(32, 8);
  ln_kernel<<<BATCH * (NSEQ / 32), lnb, 0, stream>>>(x, g, xn);
  qkv_kernel<<<(BATCH * 48 * 32) / 8, 256, 0, stream>>>(wqh, xn, qs, kT, vv);
  attn_kernel<<<(BATCH * HEADS * (NSEQ / 32)) / 8, 256, 0, stream>>>(qs, kT, vv, ao);
  proj_kernel<<<(BATCH * 16 * 32) / 8, 256, 0, stream>>>(woh, b_out, ao, out);
}
